// VAE_gru_14929306320900
// MI455X (gfx1250) — matrix operand order repaired and checked by reading, not compiled, not run on hardware
//
#include <hip/hip_runtime.h>
#include <hip/hip_bf16.h>

// ---------------------------------------------------------------------------
// Types for CDNA5 WMMA
// ---------------------------------------------------------------------------
typedef __bf16 v16bf __attribute__((ext_vector_type(16)));
typedef float  v8f   __attribute__((ext_vector_type(8)));

// ---------------------------------------------------------------------------
// Workspace layout (bytes)
// ---------------------------------------------------------------------------
static constexpr size_t OFF_PACKA = 0;                    // 3,932,160 B
static constexpr size_t OFF_GI0   = 4ull * 1024 * 1024;   // 6 MiB
static constexpr size_t OFF_H     = 10ull * 1024 * 1024;  // 3072 floats
static constexpr size_t OFF_BAR   = OFF_H + 16384;        // 4 ints

__device__ __forceinline__ float sigmoidf_(float x) {
    return 1.0f / (1.0f + __expf(-x));
}

// ---------------------------------------------------------------------------
// Kernel 1: pack encoder weights into WMMA bf16 A-fragment layout.
// ---------------------------------------------------------------------------
__global__ void pack_weights_kernel(const float* __restrict__ mu_whh,
                                    const float* __restrict__ mu_wih12,
                                    const float* __restrict__ lv_whh,
                                    const float* __restrict__ lv_wih12,
                                    v16bf* __restrict__ packA) {
    int id = blockIdx.x * blockDim.x + threadIdx.x;   // one per (slot, lane)
    if (id >= 3840 * 32) return;
    int lane = id & 31;
    int slot = id >> 5;
    int kt = slot & 7;
    int rest = slot >> 3;
    int gt = rest % 60;  rest /= 60;
    int w = rest & 3;
    int c = rest >> 2;
    int m = gt / 12, tl = gt % 12;
    int rlocal = lane & 15, hf = lane >> 4;
    int q = 16 * tl + rlocal;
    int R = (q >> 6) * 256 + 64 * w + (q & 63);
    const float* W;
    if (m < 3) W = (c ? lv_whh : mu_whh) + (size_t)m * 768 * 256;
    else       W = (c ? lv_wih12 : mu_wih12) + (size_t)(m - 3) * 768 * 256;
    const float* row = W + (size_t)R * 256 + kt * 32;
    v16bf v;
#pragma unroll
    for (int e = 0; e < 16; ++e) {
        int k = (e < 8 ? e : e + 8) + (hf ? 8 : 0);
        v[e] = (__bf16)row[k];
    }
    packA[(size_t)slot * 32 + lane] = v;
}

// ---------------------------------------------------------------------------
// Kernel 2: precompute gi0[c][t][r] = wih0[c] @ emb[idx[t]] + bih0[c]
// ---------------------------------------------------------------------------
__global__ void gi0_kernel(const int* __restrict__ mol,
                           const float* __restrict__ emb,
                           const float* __restrict__ mu_wih0,
                           const float* __restrict__ mu_bih,
                           const float* __restrict__ lv_wih0,
                           const float* __restrict__ lv_bih,
                           float* __restrict__ gi0) {
    int b = blockIdx.x;            // 0..2047
    int c = b >> 10, t = b & 1023;
    __shared__ float x[12];
    int idx = mol[t];
    if (threadIdx.x < 12) x[threadIdx.x] = emb[idx * 12 + threadIdx.x];
    __syncthreads();
    const float* W  = c ? lv_wih0 : mu_wih0;
    const float* bi = c ? lv_bih  : mu_bih;
    for (int r = threadIdx.x; r < 768; r += blockDim.x) {
        float a = bi[r];
#pragma unroll
        for (int e = 0; e < 12; ++e) a += W[(size_t)r * 12 + e] * x[e];
        gi0[((size_t)c * 1024 + t) * 768 + r] = a;
    }
}

// ---------------------------------------------------------------------------
// Kernel 3: zero h state + barriers
// ---------------------------------------------------------------------------
__global__ void init_kernel(float* __restrict__ h, int* __restrict__ bar) {
    int i = threadIdx.x;
    for (int k = i; k < 3072; k += 256) h[k] = 0.0f;
    if (i < 4) bar[i] = 0;
}

// ---------------------------------------------------------------------------
// Kernel 4: encoder. 8 workgroups x 1024 threads (see round-2 comments).
// ---------------------------------------------------------------------------
#define DO_TILE(GT, AEXPR)                                                     \
    {                                                                          \
        const int m_  = (GT) / 12, tl_ = (GT) % 12;                            \
        const int vl_ = (m_ < 3) ? m_ : (m_ - 3);                              \
        v8f acc_;                                                              \
        _Pragma("unroll") for (int g = 0; g < 8; ++g) acc_[g] = 0.0f;          \
        _Pragma("unroll") for (int kt = 0; kt < 8; ++kt) {                     \
            v16bf b_ = fragBv[(vl_ * 8 + kt) * 32 + lane];                     \
            v16bf a_ = (AEXPR);                                                \
            acc_ = __builtin_amdgcn_wmma_f32_16x16x32_bf16(                    \
                false, a_, false, b_, (short)0, acc_, false, false);           \
        }                                                                      \
        if ((lane & 15) == 0) {                                                \
            int rb_ = 16 * tl_ + lhalf * 8;                                    \
            _Pragma("unroll") for (int g = 0; g < 8; ++g)                      \
                lds_g[m_][rb_ + g] = acc_[g];                                  \
        }                                                                      \
    }

__global__ __launch_bounds__(1024) void encoder_kernel(
        const float* __restrict__ gi0,
        float* __restrict__ hglob,
        int* __restrict__ bar,
        const v16bf* __restrict__ packA,
        const float* __restrict__ bih_mu, const float* __restrict__ bhh_mu,
        const float* __restrict__ bih_lv, const float* __restrict__ bhh_lv) {
    const int tid  = threadIdx.x;
    const int wid  = tid >> 5;
    const int lane = tid & 31;
    const int lhalf = lane >> 4;
    const int c = blockIdx.x >> 2;   // chain: 0=mu, 1=lv
    const int w = blockIdx.x & 3;    // gate-column part

    const float* bih = c ? bih_lv : bih_mu;
    const float* bhh = c ? bhh_lv : bhh_mu;

    __shared__ v16bf ldsA[28 * 8 * 32];            // tile2 A-frags: 229,376 B
    __shared__ alignas(32) __bf16 fragB[3 * 8 * 32 * 16];  // B-frags: 24,576 B
    __shared__ float hstage[768];                  // per-tick h copy (3x256)
    __shared__ float lds_g[5][192];                // per-tick matvec outputs

    const v16bf* fragBv = (const v16bf*)fragB;

    // ---- resident A fragments ----
    v16bf areg[8];
    {
        const int slot1 = ((c * 4 + w) * 60 + wid) * 8;
#pragma unroll
        for (int kt = 0; kt < 8; ++kt)
            areg[kt] = packA[(size_t)(slot1 + kt) * 32 + lane];
        if (wid < 28) {
            const int slot2 = ((c * 4 + w) * 60 + 32 + wid) * 8;
#pragma unroll
            for (int kt = 0; kt < 8; ++kt)
                ldsA[(wid * 8 + kt) * 32 + lane] =
                    packA[(size_t)(slot2 + kt) * 32 + lane];
        }
    }
    // zero-fill B-fragment buffer once; only lanes 0/16 slots rewritten later.
    for (int i = tid; i < 3 * 8 * 32 * 16; i += 1024) fragB[i] = (__bf16)0.0f;
    __syncthreads();

    int* cnt = &bar[c];
    int* gen = &bar[2 + c];

    for (int tau = 0; tau < 1026; ++tau) {
        const int p = tau & 1;
        const float* hread  = hglob + ((size_t)(p * 2 + c)) * 3 * 256;
        float* hwrite = hglob + ((size_t)(((p ^ 1) * 2) + c)) * 3 * 256;

        // ---- async stage h (3 x 256 f32) global -> LDS ----
        if (tid < 768) {   // waves 0..23, wave-uniform guard
            unsigned ldsoff = (unsigned)(size_t)&hstage[tid];
            unsigned long long ga = (unsigned long long)(hread + tid);
            asm volatile("global_load_async_to_lds_b32 %0, %1, off"
                         :: "v"(ldsoff), "v"(ga) : "memory");
            asm volatile("s_wait_asynccnt 0x0" ::: "memory");
        }
        __syncthreads();
        // ---- scatter h into B-fragment layout (bf16) ----
        if (tid < 768) {
            const int vl = tid >> 8, i = tid & 255;
            const int kt = i >> 5, r = i & 31;
            const int ln = (r & 8) ? 16 : 0, e = (r & 7) + ((r >> 4) << 3);
            fragB[((vl * 8 + kt) * 32 + ln) * 16 + e] = (__bf16)hstage[tid];
        }
        __syncthreads();

        // ---- WMMA matvecs: 5 matrices x 192 rows = 60 tiles over 32 waves
        DO_TILE(wid, areg[kt]);
        if (wid < 28) {
            DO_TILE(32 + wid, ldsA[(wid * 8 + kt) * 32 + lane]);
        }
        __syncthreads();

        // ---- elementwise GRU gate combine: 3 layers x 64 elems ----
        if (tid < 192) {
            const int l = tid / 64, j = tid % 64;
            const int R0 = 64 * w + j;
            float gir, giz, gin;
            if (l == 0) {
                int t0 = (tau < 1024) ? tau : 1023;
                const float* g0 = gi0 + ((size_t)c * 1024 + t0) * 768;
                gir = g0[R0]; giz = g0[256 + R0]; gin = g0[512 + R0];
            } else {
                const int mi = 2 + l;
                gir = lds_g[mi][j]       + bih[l * 768 + R0];
                giz = lds_g[mi][64 + j]  + bih[l * 768 + 256 + R0];
                gin = lds_g[mi][128 + j] + bih[l * 768 + 512 + R0];
            }
            float ghr = lds_g[l][j]       + bhh[l * 768 + R0];
            float ghz = lds_g[l][64 + j]  + bhh[l * 768 + 256 + R0];
            float ghn = lds_g[l][128 + j] + bhh[l * 768 + 512 + R0];
            float r = sigmoidf_(gir + ghr);
            float z = sigmoidf_(giz + ghz);
            float n = tanhf(gin + r * ghn);
            float hp = hstage[l * 256 + R0];
            bool act = (tau >= l) && (tau < 1024 + l);
            float hn2 = act ? (1.0f - z) * n + z * hp : hp;
            hwrite[l * 256 + R0] = hn2;
        }
        __threadfence();
        __syncthreads();
        // ---- grid barrier across the 4 WGs of this chain ----
        if (tid == 0) {
            int g = __hip_atomic_load(gen, __ATOMIC_RELAXED,
                                      __HIP_MEMORY_SCOPE_AGENT);
            if (__hip_atomic_fetch_add(cnt, 1, __ATOMIC_ACQ_REL,
                                       __HIP_MEMORY_SCOPE_AGENT) == 3) {
                __hip_atomic_store(cnt, 0, __ATOMIC_RELAXED,
                                   __HIP_MEMORY_SCOPE_AGENT);
                __hip_atomic_fetch_add(gen, 1, __ATOMIC_RELEASE,
                                       __HIP_MEMORY_SCOPE_AGENT);
            } else {
                while (__hip_atomic_load(gen, __ATOMIC_ACQUIRE,
                                         __HIP_MEMORY_SCOPE_AGENT) == g) {
                    __builtin_amdgcn_s_sleep(1);
                }
            }
        }
        __syncthreads();
    }
}

// ---------------------------------------------------------------------------
// Kernel 5: decoder. 1 workgroup x 512 threads (16 waves).
//   d1 soft-matvec (150x256) -> WMMA: 10 M-tiles x 8 K-tiles, waves 0..9,
//     while waves 10..14 compute the 50-wide recurrent dot with VALU.
//   logits (256x50, K padded to 64) -> WMMA: 16 M-tiles x 2 K-tiles,
//     one tile per wave.
//   d2 GRU (two 150x50 dots) stays VALU.
// ---------------------------------------------------------------------------
__global__ __launch_bounds__(512) void decoder_kernel(
        const float* __restrict__ hglob,
        const float* __restrict__ eps,
        const float* __restrict__ d1_wih, const float* __restrict__ d1_whh,
        const float* __restrict__ d1_bih, const float* __restrict__ d1_bhh,
        const float* __restrict__ d2_wih, const float* __restrict__ d2_whh,
        const float* __restrict__ d2_bih, const float* __restrict__ d2_bhh,
        const float* __restrict__ w_out,  const float* __restrict__ b_out,
        float* __restrict__ out) {
    const int tid = threadIdx.x;
    const int lane = tid & 31;
    const int lhalf = lane >> 4;

    __shared__ alignas(32) __bf16 fragW1[10 * 8 * 32 * 16];  // 81,920 B
    __shared__ alignas(32) __bf16 fragWo[16 * 2 * 32 * 16];  // 32,768 B
    __shared__ alignas(32) __bf16 fragS[8 * 32 * 16];        //  8,192 B
    __shared__ alignas(32) __bf16 fragH2[2 * 32 * 16];       //  2,048 B
    __shared__ __bf16 W1h[150 * 50];
    __shared__ __bf16 W2i[150 * 50];
    __shared__ __bf16 W2h[150 * 50];
    __shared__ float dense_s[256], gi1s[152], gA[152], gB[152], cA[152],
        cB[152], h1[56], h2[56], soft[256], red[256], lgarr[256], b1h[152],
        b2i[152], b2h[152], bo[256];

    v16bf* fragW1v = (v16bf*)fragW1;
    v16bf* fragWov = (v16bf*)fragWo;
    const v16bf* fragSv  = (const v16bf*)fragS;
    const v16bf* fragH2v = (const v16bf*)fragH2;

    // ---- pack d1_wih soft-half (rows padded 150->160) into A-fragments ----
    for (int i = tid; i < 10 * 8 * 32; i += 512) {
        int ln2 = i & 31, s2 = i >> 5;
        int kt = s2 & 7, mt = s2 >> 3;
        int row = mt * 16 + (ln2 & 15);
        int hf = ln2 >> 4;
        v16bf v;
#pragma unroll
        for (int e = 0; e < 16; ++e) {
            int k = kt * 32 + (e < 8 ? e : e + 8) + (hf ? 8 : 0);
            v[e] = (row < 150) ? (__bf16)d1_wih[row * 512 + 256 + k]
                               : (__bf16)0.0f;
        }
        fragW1v[s2 * 32 + ln2] = v;
    }
    // ---- pack w_out (K padded 50->64) into A-fragments ----
    for (int i = tid; i < 16 * 2 * 32; i += 512) {
        int ln2 = i & 31, s2 = i >> 5;
        int kt = s2 & 1, mt = s2 >> 1;
        int row = mt * 16 + (ln2 & 15);
        int hf = ln2 >> 4;
        v16bf v;
#pragma unroll
        for (int e = 0; e < 16; ++e) {
            int k = kt * 32 + (e < 8 ? e : e + 8) + (hf ? 8 : 0);
            v[e] = (k < 50) ? (__bf16)w_out[row * 50 + k] : (__bf16)0.0f;
        }
        fragWov[s2 * 32 + ln2] = v;
    }
    // ---- small weights / biases / state ----
    for (int i = tid; i < 150 * 50; i += 512) {
        W1h[i] = (__bf16)d1_whh[i];
        W2i[i] = (__bf16)d2_wih[i];
        W2h[i] = (__bf16)d2_whh[i];
    }
    for (int i = tid; i < 8 * 32 * 16; i += 512) fragS[i] = (__bf16)0.0f;
    for (int i = tid; i < 2 * 32 * 16; i += 512) fragH2[i] = (__bf16)0.0f;
    if (tid < 150) {
        b1h[tid] = d1_bhh[tid];
        b2i[tid] = d2_bih[tid];
        b2h[tid] = d2_bhh[tid];
    }
    if (tid < 256) {
        bo[tid] = b_out[tid];
        soft[tid] = 0.0f;
        float mu = hglob[(size_t)(0 * 3 + 2) * 256 + tid];       // chain 0
        float lv = hglob[(size_t)(1 * 3 + 2) * 256 + tid];       // chain 1
        dense_s[tid] = mu + eps[tid] * __expf(0.1f * lv);
    }
    if (tid < 50) { h1[tid] = 0.0f; h2[tid] = 0.0f; }
    __syncthreads();
    // static gi for d1 (dense constant across steps): folds d1_bih
    if (tid < 150) {
        float a = d1_bih[tid];
        for (int k = 0; k < 256; ++k) a += d1_wih[tid * 512 + k] * dense_s[k];
        gi1s[tid] = a;
    }
    __syncthreads();

    for (int s = 0; s < 1024; ++s) {
        // ---- scatter prev_soft into B-fragment layout ----
        if (tid < 256) {
            int kt = tid >> 5, r = tid & 31;
            int ln = (r & 8) ? 16 : 0, e = (r & 7) + ((r >> 4) << 3);
            fragS[(kt * 32 + ln) * 16 + e] = (__bf16)soft[tid];
        }
        __syncthreads();
        // ---- d1 gates: WMMA soft-matvec (waves 0..9) || VALU h1-dot ----
        if (tid < 320) {
            int mt = tid >> 5;
            v8f acc;
#pragma unroll
            for (int g = 0; g < 8; ++g) acc[g] = 0.0f;
#pragma unroll
            for (int kt = 0; kt < 8; ++kt) {
                v16bf a_ = fragW1v[(mt * 8 + kt) * 32 + lane];
                v16bf b_ = fragSv[kt * 32 + lane];
                acc = __builtin_amdgcn_wmma_f32_16x16x32_bf16(
                    false, a_, false, b_, (short)0, acc, false, false);
            }
            if ((lane & 15) == 0) {
                int rb = mt * 16 + lhalf * 8;
#pragma unroll
                for (int g = 0; g < 8; ++g) {
                    int row = rb + g;
                    if (row < 150) gA[row] = acc[g] + gi1s[row];
                }
            }
        } else if (tid < 470) {
            int r = tid - 320;
            float b = b1h[r];
            const __bf16* wh = &W1h[r * 50];
            for (int k = 0; k < 50; ++k) b += (float)wh[k] * h1[k];
            gB[r] = b;
        }
        __syncthreads();
        if (tid < 50) {
            float r = sigmoidf_(gA[tid] + gB[tid]);
            float z = sigmoidf_(gA[50 + tid] + gB[50 + tid]);
            float n = tanhf(gA[100 + tid] + r * gB[100 + tid]);
            h1[tid] = (1.0f - z) * n + z * h1[tid];
        }
        __syncthreads();
        // ---- d2 GRU gates (VALU) ----
        if (tid < 150) {
            float a = b2i[tid];
            const __bf16* wi = &W2i[tid * 50];
            for (int k = 0; k < 50; ++k) a += (float)wi[k] * h1[k];
            float b = b2h[tid];
            const __bf16* wh = &W2h[tid * 50];
            for (int k = 0; k < 50; ++k) b += (float)wh[k] * h2[k];
            cA[tid] = a; cB[tid] = b;
        }
        __syncthreads();
        if (tid < 50) {
            float r = sigmoidf_(cA[tid] + cB[tid]);
            float z = sigmoidf_(cA[50 + tid] + cB[50 + tid]);
            float n = tanhf(cA[100 + tid] + r * cB[100 + tid]);
            h2[tid] = (1.0f - z) * n + z * h2[tid];
        }
        __syncthreads();
        // ---- scatter h2 into B-fragment layout (K padded to 64) ----
        if (tid < 50) {
            int kt = tid >> 5, r = tid & 31;
            int ln = (r & 8) ? 16 : 0, e = (r & 7) + ((r >> 4) << 3);
            fragH2[(kt * 32 + ln) * 16 + e] = (__bf16)h2[tid];
        }
        __syncthreads();
        // ---- logits via WMMA: 16 M-tiles, one per wave ----
        {
            int mt = tid >> 5;
            v8f acc;
#pragma unroll
            for (int g = 0; g < 8; ++g) acc[g] = 0.0f;
#pragma unroll
            for (int kt = 0; kt < 2; ++kt) {
                v16bf a_ = fragWov[(mt * 2 + kt) * 32 + lane];
                v16bf b_ = fragH2v[kt * 32 + lane];
                acc = __builtin_amdgcn_wmma_f32_16x16x32_bf16(
                    false, a_, false, b_, (short)0, acc, false, false);
            }
            if ((lane & 15) == 0) {
                int rb = mt * 16 + lhalf * 8;
#pragma unroll
                for (int g = 0; g < 8; ++g)
                    lgarr[rb + g] = acc[g] + bo[rb + g];
            }
        }
        __syncthreads();
        // ---- softmax over 256 ----
        float lg = 0.0f;
        if (tid < 256) {
            lg = lgarr[tid];
            red[tid] = lg;
        }
        __syncthreads();
        for (int st = 128; st > 0; st >>= 1) {
            if (tid < st) red[tid] = fmaxf(red[tid], red[tid + st]);
            __syncthreads();
        }
        float mx = red[0];
        __syncthreads();
        float e = __expf(lg - mx);
        if (tid < 256) red[tid] = e;
        __syncthreads();
        for (int st = 128; st > 0; st >>= 1) {
            if (tid < st) red[tid] += red[tid + st];
            __syncthreads();
        }
        float sv = e / red[0];
        __syncthreads();
        if (tid < 256) {
            soft[tid] = sv;
            out[(size_t)s * 256 + tid] = sv;
        }
        __syncthreads();
    }
}

// ---------------------------------------------------------------------------
// Host launcher
// ---------------------------------------------------------------------------
extern "C" void kernel_launch(void* const* d_in, const int* in_sizes, int n_in,
                              void* d_out, int out_size, void* d_ws,
                              size_t ws_size, hipStream_t stream) {
    (void)in_sizes; (void)n_in; (void)out_size; (void)ws_size;

    const int*   mol      = (const int*)d_in[0];
    const float* eps      = (const float*)d_in[1];
    const float* emb      = (const float*)d_in[2];
    const float* mu_wih0  = (const float*)d_in[3];
    const float* mu_wih12 = (const float*)d_in[4];
    const float* mu_whh   = (const float*)d_in[5];
    const float* mu_bih   = (const float*)d_in[6];
    const float* mu_bhh   = (const float*)d_in[7];
    const float* lv_wih0  = (const float*)d_in[8];
    const float* lv_wih12 = (const float*)d_in[9];
    const float* lv_whh   = (const float*)d_in[10];
    const float* lv_bih   = (const float*)d_in[11];
    const float* lv_bhh   = (const float*)d_in[12];
    const float* d1_wih   = (const float*)d_in[13];
    const float* d1_whh   = (const float*)d_in[14];
    const float* d1_bih   = (const float*)d_in[15];
    const float* d1_bhh   = (const float*)d_in[16];
    const float* d2_wih   = (const float*)d_in[17];
    const float* d2_whh   = (const float*)d_in[18];
    const float* d2_bih   = (const float*)d_in[19];
    const float* d2_bhh   = (const float*)d_in[20];
    const float* w_out    = (const float*)d_in[21];
    const float* b_out    = (const float*)d_in[22];

    char*  ws    = (char*)d_ws;
    v16bf* packA = (v16bf*)(ws + OFF_PACKA);
    float* gi0   = (float*)(ws + OFF_GI0);
    float* hglob = (float*)(ws + OFF_H);
    int*   bar   = (int*)(ws + OFF_BAR);

    pack_weights_kernel<<<(3840 * 32 + 255) / 256, 256, 0, stream>>>(
        mu_whh, mu_wih12, lv_whh, lv_wih12, packA);
    gi0_kernel<<<2048, 256, 0, stream>>>(mol, emb, mu_wih0, mu_bih, lv_wih0,
                                         lv_bih, gi0);
    init_kernel<<<1, 256, 0, stream>>>(hglob, bar);
    encoder_kernel<<<8, 1024, 0, stream>>>(gi0, hglob, bar, packA, mu_bih,
                                           mu_bhh, lv_bih, lv_bhh);
    decoder_kernel<<<1, 512, 0, stream>>>(hglob, eps, d1_wih, d1_whh, d1_bih,
                                          d1_bhh, d2_wih, d2_whh, d2_bih,
                                          d2_bhh, w_out, b_out, (float*)d_out);
}
